// GraphAttentionLayer_7078106104194
// MI455X (gfx1250) — compile-verified, matrix operand order checked by reading, not run on hardware
//
#include <hip/hip_runtime.h>
#include <cstdint>
#include <cstddef>

#define N_NODES 8192
#define F_IN    512
#define F_OUTC  64
#define ALPHA_F 0.2f
#define NEG_BIG_F -9.0e15f
#define LOG2E_F 1.4426950408889634f
#define CSPLIT  8
#define JCHUNK  (N_NODES / CSPLIT)      // 1024 columns per partial chunk
#define PACC_STRIDE (16 * 64 + 16)      // per (rblk,chunk): 16x64 acc + 16 rowsums

typedef float        v2f   __attribute__((ext_vector_type(2)));
typedef float        v8f   __attribute__((ext_vector_type(8)));
typedef unsigned int v8u   __attribute__((ext_vector_type(8)));
typedef __bf16       v16bf __attribute__((ext_vector_type(16)));
typedef __bf16       v2bf  __attribute__((ext_vector_type(2)));

static __device__ __forceinline__ float fast_exp2(float x) {
#if __has_builtin(__builtin_amdgcn_exp2f)
    return __builtin_amdgcn_exp2f(x);                // bare v_exp_f32
#else
    return exp2f(x);
#endif
}

static __device__ __forceinline__ unsigned int f2bf(float f) {
    unsigned int u = __float_as_uint(f);
    return (u + 0x7FFFu + ((u >> 16) & 1u)) >> 16;   // RNE; values are finite
}

#if __has_builtin(__builtin_amdgcn_cvt_pk_bf16_f32)
static __device__ __forceinline__ unsigned int pack_bf16(float x, float y) {
    union { v2bf b; unsigned int u; } cv;
    cv.b = __builtin_amdgcn_cvt_pk_bf16_f32(x, y);   // single v_cvt_pk_bf16_f32
    return cv.u;
}
#else
static __device__ __forceinline__ unsigned int pack_bf16(float x, float y) {
    return f2bf(x) | (f2bf(y) << 16);
}
#endif

// ---------------------------------------------------------------------------
// Kernel A: Wh = h @ W  (8192x512 @ 512x64) using exact f32 WMMA 16x16x4.
// 32-bit A 16x4 layout: lane L: m=L&15, hi=L>>4; VGPR0=K(2hi), VGPR1=K(2hi+1).
// B 4x16 mirrored: n=L&15; VGPR v = K(2hi+v).
// ---------------------------------------------------------------------------
__global__ __launch_bounds__(128) void wh_gemm_kernel(const float* __restrict__ h,
                                                      const float* __restrict__ W,
                                                      float* __restrict__ Wh) {
#if __has_builtin(__builtin_amdgcn_wmma_f32_16x16x4_f32)
    int lane = threadIdx.x & 31;
    int gw   = blockIdx.x * 4 + (threadIdx.x >> 5);      // 0..2047
    int mt = gw >> 2, nt = gw & 3;
    int lo = lane & 15, hi = lane >> 4;
    const float* hrow = h + (size_t)(mt * 16 + lo) * F_IN;
    const float* wcol = W + nt * 16 + lo;
    v8f acc = {};
    for (int k = 0; k < F_IN; k += 4) {
        v2f a, b;
        a.x = hrow[k + 2 * hi];
        a.y = hrow[k + 2 * hi + 1];
        b.x = wcol[(size_t)(k + 2 * hi) * F_OUTC];
        b.y = wcol[(size_t)(k + 2 * hi + 1) * F_OUTC];
        acc = __builtin_amdgcn_wmma_f32_16x16x4_f32(false, a, false, b,
                                                    (short)0, acc, false, false);
    }
#pragma unroll
    for (int i = 0; i < 8; ++i)
        Wh[(size_t)(mt * 16 + i + 8 * hi) * F_OUTC + nt * 16 + lo] = acc[i];
#else
    int idx = blockIdx.x * blockDim.x + threadIdx.x;
    for (int o = idx; o < N_NODES * F_OUTC; o += 512 * 128) {
        int row = o >> 6, col = o & 63;
        const float* hrow = h + (size_t)row * F_IN;
        float acc = 0.f;
        for (int k = 0; k < F_IN; ++k) acc += hrow[k] * W[(size_t)k * F_OUTC + col];
        Wh[o] = acc;
    }
#endif
}

// ---------------------------------------------------------------------------
// Kernel B: s1[i] = log2(e) * Wh[i,:].a1 ; s2 likewise (exp2 domain so the
// attention inner loop uses a bare v_exp_f32).
// ---------------------------------------------------------------------------
__global__ __launch_bounds__(256) void svec_kernel(const float* __restrict__ Wh,
                                                   const float* __restrict__ a,
                                                   float* __restrict__ s1,
                                                   float* __restrict__ s2) {
    int row = blockIdx.x * blockDim.x + threadIdx.x;
    if (row >= N_NODES) return;
    const float* r = Wh + (size_t)row * F_OUTC;
    float a1 = 0.f, a2 = 0.f;
#pragma unroll
    for (int n = 0; n < F_OUTC; ++n) {
        float w = r[n];
        a1 += w * a[n];
        a2 += w * a[F_OUTC + n];
    }
    s1[row] = a1 * LOG2E_F;
    s2[row] = a2 * LOG2E_F;
}

// Kernel B2: global max of (scaled) s2 — upper bound for softmax shift.
__global__ __launch_bounds__(1024) void s2max_kernel(const float* __restrict__ s2,
                                                     float* __restrict__ outmax) {
    __shared__ float red[1024];
    int tid = threadIdx.x;
    float m = -3.0e38f;
    for (int i = tid; i < N_NODES; i += 1024) m = fmaxf(m, s2[i]);
    red[tid] = m;
    __syncthreads();
    for (int s = 512; s > 0; s >>= 1) {
        if (tid < s) red[tid] = fmaxf(red[tid], red[tid + s]);
        __syncthreads();
    }
    if (tid == 0) outmax[0] = red[0];
}

// ---------------------------------------------------------------------------
// Kernel C: repack Wh (f32) into bf16 B-fragments for wmma_f32_16x16x32_bf16.
// uint index = ((jblk*4+nt)*32 + lane)*8 + v holds bf16 pair
// Wh[jblk*32 + K][nt*16 + n], K = 2v + (v>=4?8:0) + 8*hi, n = lane&15.
// Consumer loads 32B contiguous per lane (fully coalesced b128 pairs).
// ---------------------------------------------------------------------------
__global__ __launch_bounds__(256) void repack_kernel(const float* __restrict__ Wh,
                                                     unsigned int* __restrict__ WhB) {
    int idx = blockIdx.x * blockDim.x + threadIdx.x;      // 262144 total
    int v    = idx & 7;
    int lane = (idx >> 3) & 31;
    int nt   = (idx >> 8) & 3;
    int jb   = idx >> 10;                                  // 0..255
    int hi = lane >> 4, lo = lane & 15;
    int kk = 2 * v + (v >= 4 ? 8 : 0) + 8 * hi;
    int j0 = jb * 32 + kk;
    int n  = nt * 16 + lo;
    WhB[idx] = pack_bf16(Wh[(size_t)j0 * F_OUTC + n],
                         Wh[(size_t)(j0 + 1) * F_OUTC + n]);
}

// ---------------------------------------------------------------------------
// Kernel D: fused masked-softmax-numerator GEMM (partials over column chunks).
// One wave owns 16 rows x JCHUNK cols; A-fragment p-values computed in-register.
// Row sums come from a 5th WMMA against an all-ones bf16 B (co-executes with
// VALU; keeps numerator/denominator identically quantized).
// ---------------------------------------------------------------------------
__global__ __launch_bounds__(128) void attn_partial_kernel(
        const int* __restrict__ adj, const unsigned int* __restrict__ WhB,
        const float* __restrict__ s1, const float* __restrict__ s2,
        const float* __restrict__ s2max, float* __restrict__ pacc) {
    int lane = threadIdx.x & 31;
    int gw   = blockIdx.x * 4 + (threadIdx.x >> 5);       // 0..4095
    int rblk = gw / CSPLIT;
    int c    = gw % CSPLIT;
    int lo = lane & 15, hi = lane >> 4;
    int r = rblk * 16 + lo;                                // this lane's A row
    float s1r = s1[r];
    float t = s1r + s2max[0];
    float mhat = fmaxf(t, ALPHA_F * t);                    // leaky upper bound (exp2 domain)
    const int* adjrow = adj + (size_t)r * N_NODES;
    const v8u* WhB8 = (const v8u*)WhB;

    union { v8u u; v16bf b; } Bones;
#pragma unroll
    for (int v = 0; v < 8; ++v) Bones.u[v] = 0x3F803F80u;  // bf16 1.0 pairs

    v8f acc0 = {}, acc1 = {}, acc2 = {}, acc3 = {}, accS = {};

    int jend = (c + 1) * JCHUNK;
    for (int jb = c * JCHUNK; jb < jend; jb += 32) {
        int jblk = jb >> 5;
        // B fragments: contiguous 32B per lane from L2-resident packed Wh
        v8u b0 = WhB8[(size_t)(jblk * 4 + 0) * 32 + lane];
        v8u b1 = WhB8[(size_t)(jblk * 4 + 1) * 32 + lane];
        v8u b2 = WhB8[(size_t)(jblk * 4 + 2) * 32 + lane];
        v8u b3 = WhB8[(size_t)(jblk * 4 + 3) * 32 + lane];

        // adjacency + s2 for this lane's 16 K-slots: two contiguous 8-col runs
        int jc0 = jb + 8 * hi;
        __builtin_prefetch(adjrow + jc0 + 256, 0, 0);      // global_prefetch ~8 iters ahead
        int4  ad0 = *(const int4*)(adjrow + jc0);
        int4  ad1 = *(const int4*)(adjrow + jc0 + 4);
        int4  ad2 = *(const int4*)(adjrow + jc0 + 16);
        int4  ad3 = *(const int4*)(adjrow + jc0 + 20);
        float4 z0 = *(const float4*)(s2 + jc0);
        float4 z1 = *(const float4*)(s2 + jc0 + 4);
        float4 z2 = *(const float4*)(s2 + jc0 + 16);
        float4 z3 = *(const float4*)(s2 + jc0 + 20);

        auto pv = [&](int av, float zv) -> float {
            float e = s1r + zv;
            e = fmaxf(e, ALPHA_F * e);                     // leaky_relu (exp2 domain)
            e = (av > 0) ? e : NEG_BIG_F;                  // mask
            return fast_exp2(e - mhat);                    // masked -> 0
        };
        float p[16];
        p[0]=pv(ad0.x,z0.x);  p[1]=pv(ad0.y,z0.y);  p[2]=pv(ad0.z,z0.z);  p[3]=pv(ad0.w,z0.w);
        p[4]=pv(ad1.x,z1.x);  p[5]=pv(ad1.y,z1.y);  p[6]=pv(ad1.z,z1.z);  p[7]=pv(ad1.w,z1.w);
        p[8]=pv(ad2.x,z2.x);  p[9]=pv(ad2.y,z2.y);  p[10]=pv(ad2.z,z2.z); p[11]=pv(ad2.w,z2.w);
        p[12]=pv(ad3.x,z3.x); p[13]=pv(ad3.y,z3.y); p[14]=pv(ad3.z,z3.z); p[15]=pv(ad3.w,z3.w);

        union { v8u u; v16bf b; } A, B0, B1, B2, B3;
#pragma unroll
        for (int v = 0; v < 8; ++v)
            A.u[v] = pack_bf16(p[2 * v], p[2 * v + 1]);
        B0.u = b0; B1.u = b1; B2.u = b2; B3.u = b3;

        acc0 = __builtin_amdgcn_wmma_f32_16x16x32_bf16(false, A.b, false, B0.b,     (short)0, acc0, false, false);
        acc1 = __builtin_amdgcn_wmma_f32_16x16x32_bf16(false, A.b, false, B1.b,     (short)0, acc1, false, false);
        acc2 = __builtin_amdgcn_wmma_f32_16x16x32_bf16(false, A.b, false, B2.b,     (short)0, acc2, false, false);
        acc3 = __builtin_amdgcn_wmma_f32_16x16x32_bf16(false, A.b, false, B3.b,     (short)0, acc3, false, false);
        accS = __builtin_amdgcn_wmma_f32_16x16x32_bf16(false, A.b, false, Bones.b,  (short)0, accS, false, false);
    }

    float* base = pacc + (size_t)(rblk * CSPLIT + c) * PACC_STRIDE;
#pragma unroll
    for (int i = 0; i < 8; ++i) {
        int rl = i + 8 * hi;                               // local row of acc VGPR i
        base[rl * 64 + lo]      = acc0[i];
        base[rl * 64 + 16 + lo] = acc1[i];
        base[rl * 64 + 32 + lo] = acc2[i];
        base[rl * 64 + 48 + lo] = acc3[i];
        if (lo == 0) base[16 * 64 + rl] = accS[i];         // row sum (col 0 replica)
    }
}

// ---------------------------------------------------------------------------
// Kernel E: reduce column-chunk partials, divide by row sum, ELU.
// ---------------------------------------------------------------------------
__global__ __launch_bounds__(256) void finalize_kernel(const float* __restrict__ pacc,
                                                       float* __restrict__ out) {
    int idx = blockIdx.x * blockDim.x + threadIdx.x;       // 524288
    int col = idx & 63;
    int row = idx >> 6;
    int rblk = row >> 4;
    int rl   = row & 15;
    float num = 0.f, den = 0.f;
#pragma unroll
    for (int c = 0; c < CSPLIT; ++c) {
        const float* base = pacc + (size_t)(rblk * CSPLIT + c) * PACC_STRIDE;
        num += base[rl * 64 + col];
        den += base[16 * 64 + rl];
    }
    den = (den > 0.f) ? den : 1.f;
    float v = num / den;
    out[idx] = (v > 0.f) ? v : (__expf(v) - 1.f);
}

// ---------------------------------------------------------------------------
extern "C" void kernel_launch(void* const* d_in, const int* in_sizes, int n_in,
                              void* d_out, int out_size, void* d_ws, size_t ws_size,
                              hipStream_t stream) {
    const float* h   = (const float*)d_in[0];   // 8192 x 512
    const float* W   = (const float*)d_in[1];   // 512 x 64
    const float* a   = (const float*)d_in[2];   // 128 x 1
    const int*   adj = (const int*)d_in[3];     // 8192 x 8192
    float* out = (float*)d_out;

    char* ws = (char*)d_ws;
    float*        Wh   = (float*)ws;                                    // 2 MB
    unsigned int* WhB  = (unsigned int*)(ws + (size_t)2 * 1024 * 1024); // 1 MB
    float*        s1   = (float*)(ws + (size_t)3 * 1024 * 1024);        // 32 KB
    float*        s2   = s1 + N_NODES;                                  // 32 KB
    float*        s2m  = s2 + N_NODES;                                  // 4 B
    float*        pacc = (float*)(ws + (size_t)3 * 1024 * 1024 + 2 * N_NODES * 4 + 256);

    wh_gemm_kernel<<<512, 128, 0, stream>>>(h, W, Wh);
    repack_kernel<<<1024, 256, 0, stream>>>(Wh, WhB);
    svec_kernel<<<N_NODES / 256, 256, 0, stream>>>(Wh, a, s1, s2);
    s2max_kernel<<<1, 1024, 0, stream>>>(s2, s2m);
    attn_partial_kernel<<<(512 * CSPLIT) / 4, 128, 0, stream>>>(adj, WhB, s1, s2, s2m, pacc);
    finalize_kernel<<<(N_NODES * F_OUTC) / 256, 256, 0, stream>>>(pacc, out);
}
